// ObjectDetectionModelV3_83760452206680
// MI455X (gfx1250) — matrix operand order repaired and checked by reading, not compiled, not run on hardware
//
#include <hip/hip_runtime.h>
#include <hip/hip_bf16.h>

typedef __attribute__((ext_vector_type(16))) _Float16 v16h;
typedef __attribute__((ext_vector_type(8)))  float    v8f;

#define BB 4
#define NN 8
#define CC 256
#define HH 128
#define WW 128
#define DD 128
#define NEG_INF (-3.402823466e38f)

// ---------------------------------------------------------------------------
// Kernel 1: ROI max-pool. One block per (b, n) ROI; 8 waves x 32 lanes.
// Wave w owns channels [32w, 32w+32); lanes stride the ROI columns so that
// consecutive lanes hit consecutive addresses (coalesced global_load_b32).
// Wave-level max reduction via __shfl_xor (wave32). Pooled value is stored
// as f16 — it becomes the A-matrix of the WMMA GEMM in kernel 2.
// ---------------------------------------------------------------------------
__global__ void roi_pool_kernel(const float* __restrict__ encoded,
                                const float* __restrict__ bboxes,
                                _Float16* __restrict__ pooled) {
    const int roi  = blockIdx.x;           // roi = b*NN + n
    const int b    = roi / NN;
    const int lane = threadIdx.x & 31;
    const int wave = threadIdx.x >> 5;

    // ROI bounds, matching the reference's int() truncation + //2 semantics.
    const float* bb = bboxes + roi * 4;
    const int xc = (int)(bb[0] * (float)WW);
    const int yc = (int)(bb[1] * (float)HH);
    const int hw = ((int)(bb[2] * (float)WW)) >> 1;
    const int hh = ((int)(bb[3] * (float)HH)) >> 1;
    const int x0 = max(xc - hw, 0), x1 = min(xc + hw, WW - 1);  // inclusive
    const int y0 = max(yc - hh, 0), y1 = min(yc + hh, HH - 1);  // inclusive

    for (int cc = 0; cc < 32; ++cc) {
        const int c = wave * 32 + cc;
        const float* base = encoded + ((size_t)b * CC + c) * (HH * WW);
        float mv = NEG_INF;
        for (int yy = y0; yy <= y1; ++yy) {
            const float* rowp = base + yy * WW;
            if (yy < y1) __builtin_prefetch(rowp + WW + x0, 0, 0);  // global_prefetch_b8
            for (int xx = x0 + lane; xx <= x1; xx += 32) {
                mv = fmaxf(mv, rowp[xx]);
            }
        }
        // wave32 max reduction
        #pragma unroll
        for (int off = 16; off > 0; off >>= 1) {
            mv = fmaxf(mv, __shfl_xor(mv, off, 32));
        }
        if (lane == 0) {
            pooled[roi * CC + c] = (_Float16)mv;
        }
    }
}

// ---------------------------------------------------------------------------
// Kernel 2: MLP. Single block, 8 waves.
// GEMM1: h1[32,128] = pooled[32,256] @ w1^T[256,128] + b1, via
// v_wmma_f32_16x16x32_f16. Wave w owns N-columns [16w, 16w+16) and both
// M-tiles (rows 0-15 and 16-31); K loop fully unrolled (8 steps x 2 WMMAs).
// Fragment layouts per CDNA5 ISA 7.12.2:
//   A (16x32 f16): lane -> M = lane%16, half = lane/16;
//     element e (=2v+j): K = (v>=4?16:0) + half*8 + (v&3)*2 + j
//   B (32x16 f16): lane -> N = lane%16, half = lane/16; element e: same K as A
//   C/D (16x16 f32): lane -> N = lane%16; VGPR r -> M = r + (lane>=16 ? 8 : 0)
// GEMM2 (N=4) + sigmoid done with plain VALU from LDS.
// ---------------------------------------------------------------------------
__global__ void mlp_kernel(const _Float16* __restrict__ pooled,
                           const float* __restrict__ w1,
                           const float* __restrict__ b1,
                           const float* __restrict__ w2,
                           const float* __restrict__ b2,
                           float* __restrict__ out) {
    __shared__ float h1s[32][DD];   // 16 KB of the 320 KB WGP LDS

    const int tid  = threadIdx.x;
    const int lane = tid & 31;
    const int wave = tid >> 5;
    const int m    = lane & 15;     // A row / B column within tile
    const int half = lane >> 4;
    const int d0   = wave * 16;     // this wave's N-column base

    v8f c0 = {};  // rows 0..15
    v8f c1 = {};  // rows 16..31

    #pragma unroll
    for (int kt = 0; kt < CC / 32; ++kt) {
        v16h a0, a1, bf;
        #pragma unroll
        for (int e = 0; e < 16; ++e) {
            const int v  = e >> 1, j = e & 1;
            const int ka = kt * 32 + ((v >= 4) ? 16 : 0) + half * 8 + (v & 3) * 2 + j;
            const int kb = kt * 32 + ((v >= 4) ? 16 : 0) + half * 8 + (v & 3) * 2 + j;
            a0[e] = pooled[m * CC + ka];                  // A tile, rows 0..15
            a1[e] = pooled[(16 + m) * CC + ka];           // A tile, rows 16..31
            bf[e] = (_Float16)w1[(d0 + m) * CC + kb];     // B[k][n] = w1[n][k]
        }
        c0 = __builtin_amdgcn_wmma_f32_16x16x32_f16(false, a0, false, bf,
                                                    (short)0, c0, false, false);
        c1 = __builtin_amdgcn_wmma_f32_16x16x32_f16(false, a1, false, bf,
                                                    (short)0, c1, false, false);
    }

    // Scatter D tiles (+bias) to LDS.
    const int col = d0 + m;
    const float bias = b1[col];
    #pragma unroll
    for (int r = 0; r < 8; ++r) {
        const int row = r + half * 8;
        h1s[row][col]      = c0[r] + bias;
        h1s[16 + row][col] = c1[r] + bias;
    }
    __syncthreads();

    // Layer 2: out[32,4] = h1 @ w2^T + b2, then sigmoid; emit [N, B, 4].
    if (tid < 32 * 4) {
        const int row = tid >> 2;       // row = b*NN + n
        const int oc  = tid & 3;
        float acc = b2[oc];
        #pragma unroll 8
        for (int k = 0; k < DD; ++k) {
            acc += h1s[row][k] * w2[oc * DD + k];
        }
        const float sig = 1.0f / (1.0f + expf(-acc));
        const int bb = row / NN;
        const int nn = row % NN;
        out[(nn * BB + bb) * 4 + oc] = sig;
    }
}

extern "C" void kernel_launch(void* const* d_in, const int* in_sizes, int n_in,
                              void* d_out, int out_size, void* d_ws, size_t ws_size,
                              hipStream_t stream) {
    const float* encoded = (const float*)d_in[0];   // [B, C, H, W]
    const float* bboxes  = (const float*)d_in[1];   // [B, N, 4]
    const float* w1      = (const float*)d_in[2];   // [D, C]
    const float* b1      = (const float*)d_in[3];   // [D]
    const float* w2      = (const float*)d_in[4];   // [4, D]
    const float* b2      = (const float*)d_in[5];   // [4]
    float* out           = (float*)d_out;           // [N, B, 4]

    _Float16* pooled = (_Float16*)d_ws;             // [B*N, C] f16 (16 KB)

    roi_pool_kernel<<<BB * NN, 256, 0, stream>>>(encoded, bboxes, pooled);
    mlp_kernel<<<1, 256, 0, stream>>>(pooled, w1, b1, w2, b2, out);
}
